// _MambaLikeBlock_17420387353220
// MI455X (gfx1250) — matrix operand order repaired and checked by reading, not compiled, not run on hardware
//
#include <hip/hip_runtime.h>
#include <hip/hip_bf16.h>

typedef __attribute__((ext_vector_type(16))) _Float16 v16h;
typedef __attribute__((ext_vector_type(8)))  _Float16 v8h;
typedef __attribute__((ext_vector_type(8)))  float    v8f;

constexpr int L = 32768;
constexpr int C = 512;
constexpr int NCHUNK = 256;            // scan chunks
constexpr int T = L / NCHUNK;          // 128 steps per chunk

// ---------------- workspace layout (bytes) ----------------
constexpr size_t MB = 1ull << 20;
constexpr size_t OFF_HID  = 0;                     // L*C f16 (32MB); reused for h2
constexpr size_t OFF_U    = 32 * MB;               // L*C f32 (64MB)
constexpr size_t OFF_GATE = 96 * MB;               // L*C f32 (64MB); reused as ff f16 (L*2C)
constexpr size_t OFF_ST   = 160 * MB;              // L*C f16 (32MB) state
constexpr size_t OFF_X1   = 192 * MB;              // L*C f32 (64MB); fwd-scan scratch, then x1
constexpr size_t OFF_WIN  = 256 * MB;              // W_in^T  f16 (N=C x K=C)
constexpr size_t OFF_WGT  = OFF_WIN  + (size_t)C * C * 2;
constexpr size_t OFF_WOUT = OFF_WGT  + (size_t)C * C * 2;
constexpr size_t OFF_WF1  = OFF_WOUT + (size_t)C * C * 2;        // N=2C x K=C
constexpr size_t OFF_WF2  = OFF_WF1  + (size_t)2 * C * C * 2;    // N=C x K=2C
constexpr size_t OFF_CF   = OFF_WF2  + (size_t)2 * C * C * 2;    // NCHUNK*C f32 carries
constexpr size_t OFF_CB   = OFF_CF   + (size_t)NCHUNK * C * 4;
constexpr size_t OFF_INF  = OFF_CB   + (size_t)NCHUNK * C * 4;
constexpr size_t OFF_INB  = OFF_INF  + (size_t)NCHUNK * C * 4;

__device__ __forceinline__ float sigmoidf_dev(float v) {
    return 1.0f / (1.0f + __expf(-v));
}

// gfx1250 async global->LDS copy, 16B per lane (tracked by ASYNCcnt)
__device__ __forceinline__ void async_copy16(unsigned lds_addr, const void* gaddr) {
    asm volatile("global_load_async_to_lds_b128 %0, %1, off"
                 :: "v"(lds_addr), "v"(gaddr) : "memory");
}
__device__ __forceinline__ void wait_async0() {
    asm volatile("s_wait_asynccnt 0x0" ::: "memory");
}

// ---------------- weight convert + transpose: W[K][N] f32 -> Wt[N][K] f16 ----
__global__ void transpose_to_f16(const float* __restrict__ w,
                                 _Float16* __restrict__ wt, int K, int N) {
    int i = blockIdx.x * 256 + threadIdx.x;
    if (i >= K * N) return;
    int n = i / K, k = i % K;
    wt[i] = (_Float16)w[(size_t)k * N + n];
}

// ---------------- LayerNorm (row per block) -> f16 ----------------
__global__ __launch_bounds__(256)
void layernorm_to_f16(const float* __restrict__ x, const float* __restrict__ w,
                      const float* __restrict__ b, _Float16* __restrict__ out) {
    __shared__ float s1[256], s2[256];
    const int row = blockIdx.x, t = threadIdx.x;
    const size_t base = (size_t)row * C;
    const float v0 = x[base + t];
    const float v1 = x[base + t + 256];
    s1[t] = v0 + v1;
    s2[t] = v0 * v0 + v1 * v1;
    __syncthreads();
    for (int off = 128; off > 0; off >>= 1) {
        if (t < off) { s1[t] += s1[t + off]; s2[t] += s2[t + off]; }
        __syncthreads();
    }
    const float mu  = s1[0] * (1.0f / C);
    const float var = s2[0] * (1.0f / C) - mu * mu;
    const float rs  = rsqrtf(var + 1e-5f);
    out[base + t]       = (_Float16)((v0 - mu) * rs * w[t]       + b[t]);
    out[base + t + 256] = (_Float16)((v1 - mu) * rs * w[t + 256] + b[t + 256]);
}

// ---------------- WMMA GEMM: D = A(MxK,f16) x Bt(NxK,f16)^T + bias [+epilogue] ----
// MODE: 0 = f32 store, 1 = sigmoid->f32, 2 = +residual->f32, 3 = silu->f16
template <int MODE>
__global__ __launch_bounds__(256)
void gemm_wmma(const _Float16* __restrict__ A, const _Float16* __restrict__ Bt,
               const float* __restrict__ bias, const float* __restrict__ res,
               float* __restrict__ Df, _Float16* __restrict__ Dh,
               int M, int N, int K) {
    constexpr int LDS_S = 56;                       // halves per LDS row (112B, 16B mult)
    __shared__ alignas(16) _Float16 As[2][128 * LDS_S];
    __shared__ alignas(16) _Float16 Bs[2][128 * LDS_S];

    const int t    = threadIdx.x;
    const int lane = t & 31, wid = t >> 5;
    const int half = lane >> 4, l16 = lane & 15;
    const int wm   = (wid >> 2) * 64;               // wave M offset (0/64)
    const int wn   = (wid & 3) * 32;                // wave N offset (0..96)
    const int row0 = blockIdx.y * 128;
    const int col0 = blockIdx.x * 128;

    const int lr = t >> 2;                          // 0..63 loader row
    const int lc = (t & 3) * 8;                     // loader col (halves)

    // per-thread global source rows (16B granules)
    const _Float16* gA0 = A  + (size_t)(row0 + lr     ) * K + lc;
    const _Float16* gA1 = A  + (size_t)(row0 + lr + 64) * K + lc;
    const _Float16* gB0 = Bt + (size_t)(col0 + lr     ) * K + lc;
    const _Float16* gB1 = Bt + (size_t)(col0 + lr + 64) * K + lc;
    // per-thread LDS destinations (workgroup-relative byte offsets, both buffers)
    const unsigned laA0[2] = { (unsigned)(uintptr_t)&As[0][(lr     ) * LDS_S + lc],
                               (unsigned)(uintptr_t)&As[1][(lr     ) * LDS_S + lc] };
    const unsigned laA1[2] = { (unsigned)(uintptr_t)&As[0][(lr + 64) * LDS_S + lc],
                               (unsigned)(uintptr_t)&As[1][(lr + 64) * LDS_S + lc] };
    const unsigned laB0[2] = { (unsigned)(uintptr_t)&Bs[0][(lr     ) * LDS_S + lc],
                               (unsigned)(uintptr_t)&Bs[1][(lr     ) * LDS_S + lc] };
    const unsigned laB1[2] = { (unsigned)(uintptr_t)&Bs[0][(lr + 64) * LDS_S + lc],
                               (unsigned)(uintptr_t)&Bs[1][(lr + 64) * LDS_S + lc] };

    const v8f vzero = {0.f, 0.f, 0.f, 0.f, 0.f, 0.f, 0.f, 0.f};
    v8f acc[4][2];
    #pragma unroll
    for (int i = 0; i < 4; i++)
        #pragma unroll
        for (int j = 0; j < 2; j++) acc[i][j] = vzero;

    const int nk = K / 32;
    // prologue: stage tile 0 into buffer 0
    async_copy16(laA0[0], gA0);
    async_copy16(laA1[0], gA1);
    async_copy16(laB0[0], gB0);
    async_copy16(laB1[0], gB1);

    for (int i = 0; i < nk; i++) {
        wait_async0();          // this wave's pending copies (current buffer) done
        __syncthreads();        // => every wave's copies done; prev compute done
        if (i + 1 < nk) {
            const int nb = (i + 1) & 1, ko = (i + 1) * 32;
            async_copy16(laA0[nb], gA0 + ko);
            async_copy16(laA1[nb], gA1 + ko);
            async_copy16(laB0[nb], gB0 + ko);
            async_copy16(laB1[nb], gB1 + ko);
        }
        const _Float16* as = As[i & 1];
        const _Float16* bs = Bs[i & 1];

        // B fragments: lane half selects K 0..15 / 16..31, contiguous in LDS row
        v16h bfr[2];
        #pragma unroll
        for (int nt = 0; nt < 2; nt++) {
            union { v16h v; v8h h[2]; } ub;
            const int r = wn + nt * 16 + l16;
            ub.h[0] = *(const v8h*)&bs[r * LDS_S + 16 * half];
            ub.h[1] = *(const v8h*)&bs[r * LDS_S + 16 * half + 8];
            bfr[nt] = ub.v;
        }
        #pragma unroll
        for (int mt = 0; mt < 4; mt++) {
            // A fragment: lane half selects K 0..15 / 16..31, contiguous in LDS row
            union { v16h v; v8h h[2]; } ua;
            const int r = wm + mt * 16 + l16;
            ua.h[0] = *(const v8h*)&as[r * LDS_S + 16 * half];
            ua.h[1] = *(const v8h*)&as[r * LDS_S + 16 * half + 8];
            #pragma unroll
            for (int nt = 0; nt < 2; nt++) {
                acc[mt][nt] = __builtin_amdgcn_wmma_f32_16x16x32_f16(
                    false, ua.v, false, bfr[nt], (short)0, acc[mt][nt], false, false);
            }
        }
    }

    // epilogue: C/D layout -> M = half*8 + r, N = l16 within each 16x16 tile
    #pragma unroll
    for (int mt = 0; mt < 4; mt++)
        #pragma unroll
        for (int nt = 0; nt < 2; nt++) {
            const int col = col0 + wn + nt * 16 + l16;
            const float bb = bias[col];
            #pragma unroll
            for (int r = 0; r < 8; r++) {
                const int row = row0 + wm + mt * 16 + half * 8 + r;
                const size_t idx = (size_t)row * N + col;
                float v = acc[mt][nt][r] + bb;
                if constexpr (MODE == 1)      Df[idx] = sigmoidf_dev(v);
                else if constexpr (MODE == 2) Df[idx] = v + res[idx];
                else if constexpr (MODE == 3) Dh[idx] = (_Float16)(v * sigmoidf_dev(v));
                else                          Df[idx] = v;
            }
        }
}

// ---------------- scan phase 1: per-chunk fwd/bwd carries --------------------
// fwd carry: f_T = sum_t d^(T-1-t) u_t  ;  bwd carry: sum_t d^t u_t (one pass)
__global__ __launch_bounds__(256)
void scan_phase1(const float* __restrict__ u, const float* __restrict__ dl,
                 float* __restrict__ cF, float* __restrict__ cB) {
    const int c     = (blockIdx.x & 1) * 256 + threadIdx.x;
    const int chunk = blockIdx.x >> 1;
    const float d = sigmoidf_dev(dl[c]);
    const float* up = u + (size_t)chunk * T * C + c;
    float f = 0.f, b = 0.f, p = 1.f;
    #pragma unroll 4
    for (int t = 0; t < T; t++) {
        const float uv = up[(size_t)t * C];
        f = d * f + uv;
        b += p * uv;
        p *= d;
    }
    cF[chunk * C + c] = f;
    cB[chunk * C + c] = b;
}

// ---------------- scan phase 2: combine carries across chunks ----------------
__global__ __launch_bounds__(256)
void scan_phase2(const float* __restrict__ dl,
                 const float* __restrict__ cF, const float* __restrict__ cB,
                 float* __restrict__ inF, float* __restrict__ inB) {
    const int c = blockIdx.x * 256 + threadIdx.x;
    const float d  = sigmoidf_dev(dl[c]);
    const float dT = __powf(d, (float)T);
    float s = 0.f;
    for (int j = 0; j < NCHUNK; j++) { inF[j * C + c] = s; s = cF[j * C + c] + dT * s; }
    s = 0.f;
    for (int j = NCHUNK - 1; j >= 0; j--) { inB[j * C + c] = s; s = cB[j * C + c] + dT * s; }
}

// ---------------- scan phase 3: apply carries, gate, combine -> f16 ----------
__global__ __launch_bounds__(256)
void scan_phase3(const float* __restrict__ u, const float* __restrict__ gate,
                 const float* __restrict__ dl,
                 const float* __restrict__ inF, const float* __restrict__ inB,
                 float* __restrict__ fwdtmp, _Float16* __restrict__ st) {
    const int c     = (blockIdx.x & 1) * 256 + threadIdx.x;
    const int chunk = blockIdx.x >> 1;
    const float d = sigmoidf_dev(dl[c]);
    const size_t base = (size_t)chunk * T * C + c;
    float f = inF[chunk * C + c];
    #pragma unroll 4
    for (int t = 0; t < T; t++) {
        f = d * f + u[base + (size_t)t * C];
        fwdtmp[base + (size_t)t * C] = f;
    }
    float b = inB[chunk * C + c];
    #pragma unroll 4
    for (int t = T - 1; t >= 0; t--) {
        const size_t i = base + (size_t)t * C;
        b = d * b + u[i];
        st[i] = (_Float16)(0.5f * (fwdtmp[i] + b) * gate[i]);
    }
}

// ---------------- host launch ----------------
extern "C" void kernel_launch(void* const* d_in, const int* in_sizes, int n_in,
                              void* d_out, int out_size, void* d_ws, size_t ws_size,
                              hipStream_t stream) {
    const float* x    = (const float*)d_in[0];
    const float* ln1w = (const float*)d_in[1];
    const float* ln1b = (const float*)d_in[2];
    const float* W_in = (const float*)d_in[3];
    const float* b_in = (const float*)d_in[4];
    const float* W_gt = (const float*)d_in[5];
    const float* b_gt = (const float*)d_in[6];
    const float* W_o  = (const float*)d_in[7];
    const float* b_o  = (const float*)d_in[8];
    const float* dl   = (const float*)d_in[9];
    const float* ln2w = (const float*)d_in[10];
    const float* ln2b = (const float*)d_in[11];
    const float* Wf1  = (const float*)d_in[12];
    const float* bf1  = (const float*)d_in[13];
    const float* Wf2  = (const float*)d_in[14];
    const float* bf2  = (const float*)d_in[15];
    float* out = (float*)d_out;

    char* ws = (char*)d_ws;
    _Float16* hid  = (_Float16*)(ws + OFF_HID);   // also h2
    float*    u    = (float*)   (ws + OFF_U);
    float*    gate = (float*)   (ws + OFF_GATE);
    _Float16* ffh  = (_Float16*)(ws + OFF_GATE);  // reuse after scan
    _Float16* sth  = (_Float16*)(ws + OFF_ST);
    float*    x1   = (float*)   (ws + OFF_X1);    // fwd scratch, then x + state@W_out
    _Float16* tWin = (_Float16*)(ws + OFF_WIN);
    _Float16* tWgt = (_Float16*)(ws + OFF_WGT);
    _Float16* tWo  = (_Float16*)(ws + OFF_WOUT);
    _Float16* tWf1 = (_Float16*)(ws + OFF_WF1);
    _Float16* tWf2 = (_Float16*)(ws + OFF_WF2);
    float* cF  = (float*)(ws + OFF_CF);
    float* cB  = (float*)(ws + OFF_CB);
    float* inF = (float*)(ws + OFF_INF);
    float* inB = (float*)(ws + OFF_INB);

    // 1) weights -> f16, transposed to [N][K]
    transpose_to_f16<<<(C * C + 255) / 256, 256, 0, stream>>>(W_in, tWin, C, C);
    transpose_to_f16<<<(C * C + 255) / 256, 256, 0, stream>>>(W_gt, tWgt, C, C);
    transpose_to_f16<<<(C * C + 255) / 256, 256, 0, stream>>>(W_o,  tWo,  C, C);
    transpose_to_f16<<<(C * 2 * C + 255) / 256, 256, 0, stream>>>(Wf1, tWf1, C, 2 * C);
    transpose_to_f16<<<(2 * C * C + 255) / 256, 256, 0, stream>>>(Wf2, tWf2, 2 * C, C);

    // 2) ln1
    layernorm_to_f16<<<L, 256, 0, stream>>>(x, ln1w, ln1b, hid);

    // 3) u = hid @ W_in + b_in ; 4) gate = sigmoid(hid @ W_gate + b_gate)
    dim3 g512(C / 128, L / 128);
    gemm_wmma<0><<<g512, 256, 0, stream>>>(hid, tWin, b_in, nullptr, u,  nullptr, L, C, C);
    gemm_wmma<1><<<g512, 256, 0, stream>>>(hid, tWgt, b_gt, nullptr, gate, nullptr, L, C, C);

    // 5-7) bidirectional chunked scan -> state f16
    scan_phase1<<<NCHUNK * 2, 256, 0, stream>>>(u, dl, cF, cB);
    scan_phase2<<<2, 256, 0, stream>>>(dl, cF, cB, inF, inB);
    scan_phase3<<<NCHUNK * 2, 256, 0, stream>>>(u, gate, dl, inF, inB, x1, sth);

    // 8) x1 = x + state @ W_out + b_out
    gemm_wmma<2><<<g512, 256, 0, stream>>>(sth, tWo, b_o, x, x1, nullptr, L, C, C);

    // 9) ln2 -> h2 (reuse hid)
    layernorm_to_f16<<<L, 256, 0, stream>>>(x1, ln2w, ln2b, hid);

    // 10) ff = silu(h2 @ W_ff1 + b_ff1) f16
    dim3 g1024(2 * C / 128, L / 128);
    gemm_wmma<3><<<g1024, 256, 0, stream>>>(hid, tWf1, bf1, nullptr, nullptr, ffh, L, 2 * C, C);

    // 11) out = x1 + ff @ W_ff2 + b_ff2
    gemm_wmma<2><<<g512, 256, 0, stream>>>(ffh, tWf2, bf2, x1, out, nullptr, L, C, 2 * C);
}
